// Rvt_80865644249200
// MI455X (gfx1250) — matrix operand order repaired and checked by reading, not compiled, not run on hardware
//
#include <hip/hip_runtime.h>

// ---------------------------------------------------------------------------
// MI455X (gfx1250): bf16 WMMA everywhere, fp32 accumulate.
// Pipeline: prep(bf16 casts/packs) -> proj GEMM -> grouped convs (q, fused kv)
//           -> attention (LDS-resident softmax) -> fused gates + LSTM cell.
// All GEMM loops: double-buffered LDS tiles, b128 staging, v8bf im2col writes.
// ---------------------------------------------------------------------------

typedef __bf16 bf16;
typedef bf16  v8bf  __attribute__((ext_vector_type(8)));
typedef bf16  v16bf __attribute__((ext_vector_type(16)));
typedef float v8f   __attribute__((ext_vector_type(8)));

constexpr int N_  = 4;
constexpr int I_  = 64;
constexpr int C_  = 256;
constexpr int G_  = 8;
constexpr int Kk  = 9;
constexpr int H_  = 48;
constexpr int W_  = 48;
constexpr int Cg  = 32;                // C/G
constexpr int HW  = H_ * W_;           // 2304 (q pixels)
constexpr int OD  = 40;                // H-K+1
constexpr int D_  = OD * OD;           // 1600 (k/v pixels)
constexpr int KK  = 2 * Cg * Kk * Kk;  // 5184 (grouped-conv K dim)
constexpr int KG  = KK + Cg;           // 5216 (gates: conv + 1x1 Wa term)

#define DEV static __device__ __forceinline__

DEV bf16 f2bf(float f) {
  unsigned u = __builtin_bit_cast(unsigned, f);
  u += 0x7fffu + ((u >> 16) & 1u);
  unsigned short h = (unsigned short)(u >> 16);
  return __builtin_bit_cast(bf16, h);
}
DEV bf16 bf_zero() { return __builtin_bit_cast(bf16, (unsigned short)0); }

DEV v16bf catbf(v8bf lo, v8bf hi) {
  return __builtin_shufflevector(lo, hi, 0,1,2,3,4,5,6,7,8,9,10,11,12,13,14,15);
}
DEV v8f wmma_bf16(v16bf a, v16bf b, v8f c) {
  return __builtin_amdgcn_wmma_f32_16x16x32_bf16(false, a, false, b,
                                                 (short)0, c, false, false);
}
DEV v8f vzero8() { v8f z = {0.f,0.f,0.f,0.f,0.f,0.f,0.f,0.f}; return z; }

// A-fragment (16x32 bf16, MxK) from row-major (stride ldh halfs).
// Lane l: row = l&15, K halves {(l>>4)*8..+7} and {16+(l>>4)*8..+7}
DEV v16bf load_afrag(const bf16* base, int lane, int ldh) {
  int m = lane & 15, q = lane >> 4;
  v8bf lo = *(const v8bf*)(base + (size_t)m * ldh + q * 8);
  v8bf hi = *(const v8bf*)(base + (size_t)m * ldh + 16 + q * 8);
  return catbf(lo, hi);
}
// B-fragment (32x16 bf16, KxN) col-major (col stride ldh halfs, K contiguous).
// Lane l: col = l&15, K halves {(l>>4)*8..+7} and {16+(l>>4)*8..+7}
DEV v16bf load_bfrag(const bf16* base, int lane, int ldh) {
  int c = lane & 15, q = lane >> 4;
  v8bf lo = *(const v8bf*)(base + (size_t)c * ldh + q * 8);
  v8bf hi = *(const v8bf*)(base + (size_t)c * ldh + 16 + q * 8);
  return catbf(lo, hi);
}

// im2col chunk: thread `tid` fills Bs[p][kc8..kc8+7] for pixel p = tid>>2,
// 8 consecutive K values, one address decomposition + incremental rollover,
// single b128 LDS store. PAD=0 (VALID) folds away all bounds checks.
template <int PAD, int OWd>
DEV void im2col_fill(bf16* Bs, const bf16* xhg, int pbase, int k0, int tid) {
  int p = tid >> 2, kc8 = (tid & 3) * 8;
  int kk = k0 + kc8;
  int ci = kk / 81;
  int r = kk - ci * 81;
  int ky = r / 9, kx = r - ky * 9;
  int pix = pbase + p;
  int oy = pix / OWd, ox = pix - oy * OWd;
  int iy = oy + ky - PAD;
  const bf16* src = xhg + (size_t)ci * HW + (ptrdiff_t)iy * W_ + (ox - PAD);
  v8bf vals;
#pragma unroll
  for (int j = 0; j < 8; ++j) {
    int ix = ox + kx - PAD;
    bf16 val = bf_zero();
    if (PAD == 0 || (iy >= 0 && iy < H_ && ix >= 0 && ix < W_)) val = src[kx];
    vals[j] = val;
    ++kx;
    if (kx == 9) {
      kx = 0; ++ky; ++iy; src += W_;
      if (ky == 9) { ky = 0; iy -= 9; src += (ptrdiff_t)HW - 9 * W_; }
    }
  }
  *(v8bf*)&Bs[p * 32 + kc8] = vals;
}

// ---------------------------------------------------------------------------
// Prep kernels
// ---------------------------------------------------------------------------
__global__ void k_cvt(const float* __restrict__ s, bf16* __restrict__ d, int n) {
  int i = blockIdx.x * 256 + threadIdx.x;
  if (i < n) d[i] = f2bf(s[i]);
}

__global__ void k_prep_wproj(const float* __restrict__ Wx,
                             const float* __restrict__ Wxg,
                             bf16* __restrict__ wA) {
  int e = blockIdx.x * 256 + threadIdx.x;
  if (e >= C_ * 320) return;
  int m = e / 320, c = e - m * 320;
  float v = (c < I_) ? Wx[m * I_ + c] : Wxg[m * C_ + (c - I_)];
  wA[e] = f2bf(v);
}

// Stacked K|V weights: wKV[g][64][KK], rows 0-31 = Wk group rows, 32-63 = Wv
__global__ void k_prep_wkv(const float* __restrict__ Wk,
                           const float* __restrict__ Wv,
                           bf16* __restrict__ wKV) {
  int e = blockIdx.x * 256 + threadIdx.x;
  if (e >= 2 * C_ * KK) return;
  int half = e / (C_ * KK);
  int r = e - half * C_ * KK;
  int c = r / KK, col = r - c * KK;
  int g = c >> 5, co = c & 31;
  const float* src = half ? Wv : Wk;
  wKV[((size_t)(g * 64 + half * 32 + co)) * KK + col] = f2bf(src[(size_t)c * KK + col]);
}

__global__ void k_prep_wgate(const float* __restrict__ Wxh0, const float* __restrict__ Wxh1,
                             const float* __restrict__ Wxh2, const float* __restrict__ Wxh3,
                             const float* __restrict__ Wa0,  const float* __restrict__ Wa1,
                             const float* __restrict__ Wa2,  const float* __restrict__ Wa3,
                             bf16* __restrict__ wG) {
  int e = blockIdx.x * 256 + threadIdx.x;
  if (e >= G_ * 128 * KG) return;
  int g = e / (128 * KG);
  int r = (e / KG) % 128;
  int col = e % KG;
  int gate = r >> 5, co = r & 31, c = g * 32 + co;
  const float* Wxh[4] = {Wxh0, Wxh1, Wxh2, Wxh3};
  const float* Wa[4]  = {Wa0, Wa1, Wa2, Wa3};
  float v = (col < KK) ? Wxh[gate][(size_t)c * KK + col]
                       : Wa[gate][c * Cg + (col - KK)];
  wG[e] = f2bf(v);
}

__global__ void k_pack_bproj(const float* __restrict__ inp,
                             const float* __restrict__ h,
                             bf16* __restrict__ Bp) {
  int e = blockIdx.x * 256 + threadIdx.x;
  if (e >= N_ * HW * 320) return;
  int p = e / 320, c = e - p * 320;
  int n = p / HW, pix = p - n * HW;
  float v = (c < I_) ? inp[((size_t)(n * I_ + c)) * HW + pix]
                     : h[((size_t)(n * C_ + (c - I_))) * HW + pix];
  Bp[e] = f2bf(v);
}

__global__ void k_pack_xhh(const float* __restrict__ h, bf16* __restrict__ xh) {
  int e = blockIdx.x * 256 + threadIdx.x;
  if (e >= N_ * C_ * HW) return;
  int n = e / (C_ * HW);
  int r = e - n * C_ * HW;
  int c = r / HW, pix = r - c * HW;
  int g = c >> 5, ci = c & 31;
  xh[((size_t)(n * 512 + g * 64 + 32 + ci)) * HW + pix] = f2bf(h[e]);
}

// ---------------------------------------------------------------------------
// Projection GEMM: x = [Wx|Wxg] * [inp;h]  (M=256, K=320, Npix=9216)
// 64x64 block tile, double-buffered b128 staging, writes x-half of xh.
// ---------------------------------------------------------------------------
__global__ __launch_bounds__(256) void k_proj(const bf16* __restrict__ A,
                                              const bf16* __restrict__ B,
                                              bf16* __restrict__ xh) {
  int bm = blockIdx.x & 3, bn = blockIdx.x >> 2;
  int tid = threadIdx.x, lane = tid & 31, wave = tid >> 5;
  int nt = wave & 3, mtb = (wave >> 2) * 2;
  __shared__ __align__(64) bf16 As[2][64 * 32];
  __shared__ __align__(64) bf16 Bs[2][64 * 32];
  v8f acc0 = vzero8(), acc1 = vzero8();

  int row = tid >> 2, cc = tid & 3;
  auto fill = [&](int k0, int buf) {
    *(v8bf*)&As[buf][row * 32 + cc * 8] =
        *(const v8bf*)&A[(size_t)(bm * 64 + row) * 320 + k0 + cc * 8];
    *(v8bf*)&Bs[buf][row * 32 + cc * 8] =
        *(const v8bf*)&B[(size_t)(bn * 64 + row) * 320 + k0 + cc * 8];
  };

  fill(0, 0);
  for (int ks = 0; ks < 10; ++ks) {
    __syncthreads();
    int b = ks & 1;
    v16bf bfr = load_bfrag(&Bs[b][(nt * 16) * 32], lane, 32);
    v16bf a0  = load_afrag(&As[b][(mtb * 16) * 32], lane, 32);
    v16bf a1  = load_afrag(&As[b][((mtb + 1) * 16) * 32], lane, 32);
    acc0 = wmma_bf16(a0, bfr, acc0);
    acc1 = wmma_bf16(a1, bfr, acc1);
    if (ks + 1 < 10) fill((ks + 1) * 32, (ks + 1) & 1);
  }
  int hi = lane >> 4, col = lane & 15;
  int pixg = bn * 64 + nt * 16 + col;
  int n = pixg / HW, pix = pixg - n * HW;
  for (int i = 0; i < 2; ++i) {
    v8f acc = i ? acc1 : acc0;
    for (int x = 0; x < 8; ++x) {
      int m = bm * 64 + (mtb + i) * 16 + hi * 8 + x;
      int g = m >> 5, ci = m & 31;
      xh[((size_t)(n * 512 + g * 64 + ci)) * HW + pix] = f2bf(acc[x]);
    }
  }
}

// ---------------------------------------------------------------------------
// q conv (SAME): implicit GEMM, tile 32ch x 64px, 8 waves = 2mt x 4nt.
// Output pixel-major q[(ng*HW+pix)*32+ch] bf16.
// ---------------------------------------------------------------------------
__global__ __launch_bounds__(256) void k_gconv_q(const bf16* __restrict__ Wt,
                                                 const bf16* __restrict__ xh,
                                                 bf16* __restrict__ out) {
  constexpr int NT = HW / 64;
  int bx = blockIdx.x;
  int tile = bx % NT;
  int g = (bx / NT) & 7;
  int n = bx / (NT * 8);
  int pbase = tile * 64;
  int tid = threadIdx.x, lane = tid & 31, wave = tid >> 5;
  int mt = wave >> 2, nt = wave & 3;
  const bf16* xhg = xh + (size_t)(n * 512 + g * 64) * HW;
  const bf16* Wg  = Wt + (size_t)(g * Cg) * KK;
  __shared__ __align__(64) bf16 As[2][Cg * 32];
  __shared__ __align__(64) bf16 Bs[2][64 * 32];
  v8f acc = vzero8();

  auto fill = [&](int k0, int buf) {
    if (tid < 128) {
      int row = tid >> 2, cc = tid & 3;
      *(v8bf*)&As[buf][row * 32 + cc * 8] =
          *(const v8bf*)&Wg[(size_t)row * KK + k0 + cc * 8];
    }
    im2col_fill<4, W_>(Bs[buf], xhg, pbase, k0, tid);
  };

  constexpr int NSTEP = KK / 32;  // 162
  fill(0, 0);
  for (int ks = 0; ks < NSTEP; ++ks) {
    __syncthreads();
    int b = ks & 1;
    v16bf afr = load_afrag(&As[b][(mt * 16) * 32], lane, 32);
    v16bf bfr = load_bfrag(&Bs[b][(nt * 16) * 32], lane, 32);
    acc = wmma_bf16(afr, bfr, acc);
    if (ks + 1 < NSTEP) fill((ks + 1) * 32, (ks + 1) & 1);
  }
  int hi = lane >> 4, col = lane & 15;
  int pix = pbase + nt * 16 + col;
  size_t ng = (size_t)(n * 8 + g);
  v8bf o;
  for (int x = 0; x < 8; ++x) o[x] = f2bf(acc[x]);
  *(v8bf*)&out[(ng * HW + pix) * 32 + mt * 16 + hi * 8] = o;
}

// ---------------------------------------------------------------------------
// Fused k+v conv (VALID): A = stacked [64 rows = k(32)|v(32)][5184] per group,
// one im2col gather feeds 2 WMMAs/wave/k-step. k -> pixel-major, v -> ch-major.
// ---------------------------------------------------------------------------
__global__ __launch_bounds__(256) void k_gconv_kv(const bf16* __restrict__ Wkv,
                                                  const bf16* __restrict__ xh,
                                                  bf16* __restrict__ outK,
                                                  bf16* __restrict__ outV) {
  constexpr int NT = D_ / 64;  // 25
  int bx = blockIdx.x;
  int tile = bx % NT;
  int g = (bx / NT) & 7;
  int n = bx / (NT * 8);
  int pbase = tile * 64;
  int tid = threadIdx.x, lane = tid & 31, wave = tid >> 5;
  int nt = wave & 3, mtb = (wave >> 2) * 2;
  const bf16* xhg = xh + (size_t)(n * 512 + g * 64) * HW;
  const bf16* Wg  = Wkv + (size_t)(g * 64) * KK;
  __shared__ __align__(64) bf16 As[2][64 * 32];
  __shared__ __align__(64) bf16 Bs[2][64 * 32];
  v8f acc0 = vzero8(), acc1 = vzero8();

  int row = tid >> 2, cc = tid & 3;
  auto fill = [&](int k0, int buf) {
    *(v8bf*)&As[buf][row * 32 + cc * 8] =
        *(const v8bf*)&Wg[(size_t)row * KK + k0 + cc * 8];
    im2col_fill<0, OD>(Bs[buf], xhg, pbase, k0, tid);
  };

  constexpr int NSTEP = KK / 32;
  fill(0, 0);
  for (int ks = 0; ks < NSTEP; ++ks) {
    __syncthreads();
    int b = ks & 1;
    v16bf bfr = load_bfrag(&Bs[b][(nt * 16) * 32], lane, 32);
    v16bf a0  = load_afrag(&As[b][(mtb * 16) * 32], lane, 32);
    v16bf a1  = load_afrag(&As[b][((mtb + 1) * 16) * 32], lane, 32);
    acc0 = wmma_bf16(a0, bfr, acc0);
    acc1 = wmma_bf16(a1, bfr, acc1);
    if (ks + 1 < NSTEP) fill((ks + 1) * 32, (ks + 1) & 1);
  }
  int hi = lane >> 4, col = lane & 15;
  int pix = pbase + nt * 16 + col;
  size_t ng = (size_t)(n * 8 + g);
  for (int i = 0; i < 2; ++i) {
    int mtg = mtb + i;
    v8f acc = i ? acc1 : acc0;
    if (mtg < 2) {  // k rows: pixel-major
      v8bf o;
      for (int x = 0; x < 8; ++x) o[x] = f2bf(acc[x]);
      *(v8bf*)&outK[(ng * D_ + pix) * 32 + mtg * 16 + hi * 8] = o;
    } else {        // v rows: channel-major
      for (int x = 0; x < 8; ++x) {
        int ch = (mtg - 2) * 16 + hi * 8 + x;
        outV[(ng * Cg + ch) * D_ + pix] = f2bf(acc[x]);
      }
    }
  }
}

// ---------------------------------------------------------------------------
// Attention: per (n,g,16-query tile). S fp32 in LDS, two-pass softmax ->
// P bf16 in LDS, O = v * P^T with split-K over waves + LDS reduce.
// ---------------------------------------------------------------------------
constexpr int ATTN_SMEM = 16 * D_ * 4 + 16 * D_ * 2 + 8 * 256 * 4 + 16 * 16 * 4 + 16 * 4;

__global__ __launch_bounds__(256) void k_attn(const bf16* __restrict__ q,
                                              const bf16* __restrict__ k,
                                              const bf16* __restrict__ v,
                                              const float* __restrict__ tau,
                                              bf16* __restrict__ a) {
  int bx = blockIdx.x;
  int qt = bx % (HW / 16);
  int g  = (bx / (HW / 16)) & 7;
  int n  = bx / ((HW / 16) * 8);
  int ng = n * 8 + g;
  int qbase = qt * 16;
  int tid = threadIdx.x, lane = tid & 31, wave = tid >> 5;

  extern __shared__ char smem[];
  float* S    = (float*)smem;                  // [16][1600] f32
  bf16*  P    = (bf16*)(smem + 16 * D_ * 4);   // [16][1600] bf16
  float* Op   = (float*)(smem + 16 * D_ * 6);  // [8][256] split-K partials
  float* red  = Op + 8 * 256;                  // [16][16]
  float* ssum = red + 256;                     // [16]
  float tauv = tau[g];

  // Phase 1: S = tau * q^T k
  const bf16* qb = q + (size_t)(ng * HW + qbase) * 32;
  v16bf afr = load_afrag(qb, lane, 32);
  const bf16* kb = k + (size_t)ng * D_ * 32;
  for (int dt = wave; dt < D_ / 16; dt += 8) {
    v16bf bfr = load_bfrag(kb + (size_t)(dt * 16) * 32, lane, 32);
    v8f s = vzero8();
    s = wmma_bf16(afr, bfr, s);
    int key = dt * 16 + (lane & 15);
    int qrow = (lane >> 4) * 8;
    for (int x = 0; x < 8; ++x) S[(qrow + x) * D_ + key] = s[x] * tauv;
  }
  __syncthreads();

  // Phase 2: softmax per row (16 threads per row)
  int row = tid >> 4, sub = tid & 15;
  float* Sr = S + row * D_;
  float mx = -3.0e38f;
  for (int d = sub; d < D_; d += 16) mx = fmaxf(mx, Sr[d]);
  red[row * 16 + sub] = mx;
  __syncthreads();
  mx = -3.0e38f;
  for (int t = 0; t < 16; ++t) mx = fmaxf(mx, red[row * 16 + t]);
  __syncthreads();
  float sm = 0.f;
  bf16* Pr = P + row * D_;
  for (int d = sub; d < D_; d += 16) {
    float e = __expf(Sr[d] - mx);
    sm += e;
    Pr[d] = f2bf(e);  // unnormalized; divide at epilogue
  }
  red[row * 16 + sub] = sm;
  __syncthreads();
  if (tid < 16) {
    float s2 = 0.f;
    for (int t = 0; t < 16; ++t) s2 += red[tid * 16 + t];
    ssum[tid] = s2;
  }
  __syncthreads();

  // Phase 3: O = v * P^T (split-K across wave pairs)
  int sb = wave & 1, kq = wave >> 1;
  const bf16* vb = v + (size_t)ng * Cg * D_ + (size_t)(sb * 16) * D_;
  v8f oacc = vzero8();
  for (int j = kq; j < D_ / 32; j += 4) {
    int d0 = j * 32;
    int ch = lane & 15, qh = lane >> 4;
    v8bf lo = *(const v8bf*)&vb[(size_t)ch * D_ + d0 + qh * 8];
    v8bf hi = *(const v8bf*)&vb[(size_t)ch * D_ + d0 + 16 + qh * 8];
    v16bf va = catbf(lo, hi);
    int qc = lane & 15;
    v8bf plo = *(const v8bf*)&P[(size_t)qc * D_ + d0 + qh * 8];
    v8bf phi = *(const v8bf*)&P[(size_t)qc * D_ + d0 + 16 + qh * 8];
    oacc = wmma_bf16(va, catbf(plo, phi), oacc);
  }
  for (int x = 0; x < 8; ++x) Op[wave * 256 + x * 32 + lane] = oacc[x];
  __syncthreads();

  for (int t = tid; t < 512; t += 256) {
    int sb2 = t >> 8, e = t & 255, vi = e >> 5, l = e & 31;
    float val = Op[(0 * 2 + sb2) * 256 + e] + Op[(1 * 2 + sb2) * 256 + e] +
                Op[(2 * 2 + sb2) * 256 + e] + Op[(3 * 2 + sb2) * 256 + e];
    int qloc = l & 15;
    int ch = sb2 * 16 + vi + 8 * (l >> 4);
    val /= ssum[qloc];
    a[((size_t)(ng * HW + qbase + qloc)) * 32 + ch] = f2bf(val);
  }
}

// ---------------------------------------------------------------------------
// Fused gates + LSTM cell. A = per-group [128 = 4 gates x 32ch][5216]:
// cols 0..5183 = Wxh (im2col of xh), 5184..5215 = Wa (1x1 of a).
// One shared B tile feeds 4 WMMAs/wave/k-step. Epilogue: bias + activations
// + cell update -> h_new (fp32) to d_out.
// ---------------------------------------------------------------------------
__global__ __launch_bounds__(256) void k_gates(const bf16* __restrict__ Wg8,
                                               const bf16* __restrict__ xh,
                                               const bf16* __restrict__ a,
                                               const float* __restrict__ c_prev,
                                               const float* __restrict__ bi,
                                               const float* __restrict__ bff,
                                               const float* __restrict__ bg,
                                               const float* __restrict__ bo,
                                               float* __restrict__ out) {
  int bx = blockIdx.x;
  int tile = bx % 36;
  int g = (bx / 36) & 7;
  int n = bx / (36 * 8);
  int pbase = tile * 64;
  int ng = n * 8 + g;
  int tid = threadIdx.x, lane = tid & 31, wave = tid >> 5;
  int nt = wave & 3, mtb = (wave >> 2) * 4;
  const bf16* xhg  = xh + (size_t)(n * 512 + g * 64) * HW;
  const bf16* Wgrp = Wg8 + (size_t)g * 128 * KG;
  __shared__ __align__(64) bf16 As[2][128 * 32];
  __shared__ __align__(64) bf16 Bs[2][64 * 32];
  __shared__ float gbuf[128 * 64];
  v8f acc[4];
  for (int i = 0; i < 4; ++i) acc[i] = vzero8();

  auto fill = [&](int k0, int buf) {
    for (int t = tid; t < 512; t += 256) {
      int row = t >> 2, cc = t & 3;
      *(v8bf*)&As[buf][row * 32 + cc * 8] =
          *(const v8bf*)&Wgrp[(size_t)row * KG + k0 + cc * 8];
    }
    if (k0 < KK) {
      im2col_fill<4, W_>(Bs[buf], xhg, pbase, k0, tid);
    } else {  // final step: B = a (1x1 Wa term), contiguous b128 copy
      int p = tid >> 2, cc2 = tid & 3;
      *(v8bf*)&Bs[buf][p * 32 + cc2 * 8] =
          *(const v8bf*)&a[((size_t)(ng * HW + pbase + p)) * 32 + cc2 * 8];
    }
  };

  constexpr int NSTEP = KG / 32;  // 163
  fill(0, 0);
  for (int ks = 0; ks < NSTEP; ++ks) {
    __syncthreads();
    int b = ks & 1;
    v16bf bfr = load_bfrag(&Bs[b][(nt * 16) * 32], lane, 32);
    for (int i = 0; i < 4; ++i) {
      v16bf afr = load_afrag(&As[b][((mtb + i) * 16) * 32], lane, 32);
      acc[i] = wmma_bf16(afr, bfr, acc[i]);
    }
    if (ks + 1 < NSTEP) fill((ks + 1) * 32, (ks + 1) & 1);
  }

  int hi = lane >> 4, col = lane & 15;
  for (int i = 0; i < 4; ++i)
    for (int x = 0; x < 8; ++x)
      gbuf[((mtb + i) * 16 + hi * 8 + x) * 64 + nt * 16 + col] = acc[i][x];
  __syncthreads();
  for (int t = tid; t < Cg * 64; t += 256) {
    int co = t >> 6, p = t & 63;
    int c = g * 32 + co;
    int pix = pbase + p;
    float pi = gbuf[(0 * 32 + co) * 64 + p] + bi[c];
    float pf = gbuf[(1 * 32 + co) * 64 + p] + bff[c];
    float pg = gbuf[(2 * 32 + co) * 64 + p] + bg[c];
    float po = gbuf[(3 * 32 + co) * 64 + p] + bo[c];
    float iv = 1.f / (1.f + __expf(-pi));
    float fv = 1.f / (1.f + __expf(-pf));
    float gv = tanhf(pg);
    float ov = 1.f / (1.f + __expf(-po));
    float cp = c_prev[((size_t)(n * C_ + c)) * HW + pix];
    float cn = fv * cp + iv * gv;
    out[((size_t)(n * C_ + c)) * HW + pix] = ov * tanhf(cn);
  }
}

// ---------------------------------------------------------------------------
// Workspace layout (bytes)
// ---------------------------------------------------------------------------
constexpr size_t OFF_WA  = 0;
constexpr size_t OFF_BP  = OFF_WA  + (size_t)C_ * 320 * 2;
constexpr size_t OFF_XH  = OFF_BP  + (size_t)N_ * HW * 320 * 2;
constexpr size_t OFF_WQ  = OFF_XH  + (size_t)N_ * 512 * HW * 2;
constexpr size_t OFF_WKV = OFF_WQ  + (size_t)C_ * KK * 2;
constexpr size_t OFF_WG  = OFF_WKV + (size_t)2 * C_ * KK * 2;
constexpr size_t OFF_Q   = OFF_WG  + (size_t)G_ * 128 * KG * 2;
constexpr size_t OFF_K   = OFF_Q   + (size_t)N_ * G_ * HW * 32 * 2;
constexpr size_t OFF_V   = OFF_K   + (size_t)N_ * G_ * D_ * 32 * 2;
constexpr size_t OFF_A   = OFF_V   + (size_t)N_ * G_ * Cg * D_ * 2;

extern "C" void kernel_launch(void* const* d_in, const int* in_sizes, int n_in,
                              void* d_out, int out_size, void* d_ws, size_t ws_size,
                              hipStream_t stream) {
  (void)in_sizes; (void)n_in; (void)out_size; (void)ws_size;
  const float* inp    = (const float*)d_in[0];
  const float* h_prev = (const float*)d_in[1];
  const float* c_prev = (const float*)d_in[2];
  const float* Wx     = (const float*)d_in[3];
  const float* Wxg    = (const float*)d_in[4];
  const float* tau    = (const float*)d_in[5];
  const float* Wq     = (const float*)d_in[6];
  const float* Wk     = (const float*)d_in[7];
  const float* Wv     = (const float*)d_in[8];
  const float* Wa_i   = (const float*)d_in[9];
  const float* Wxh_i  = (const float*)d_in[10];
  const float* b_i    = (const float*)d_in[11];
  const float* Wa_f   = (const float*)d_in[12];
  const float* Wxh_f  = (const float*)d_in[13];
  const float* b_f    = (const float*)d_in[14];
  const float* Wa_g   = (const float*)d_in[15];
  const float* Wxh_g  = (const float*)d_in[16];
  const float* b_g    = (const float*)d_in[17];
  const float* Wa_o   = (const float*)d_in[18];
  const float* Wxh_o  = (const float*)d_in[19];
  const float* b_o    = (const float*)d_in[20];

  char* ws = (char*)d_ws;
  bf16* wA  = (bf16*)(ws + OFF_WA);
  bf16* Bp  = (bf16*)(ws + OFF_BP);
  bf16* xh  = (bf16*)(ws + OFF_XH);
  bf16* wQ  = (bf16*)(ws + OFF_WQ);
  bf16* wKV = (bf16*)(ws + OFF_WKV);
  bf16* wG  = (bf16*)(ws + OFF_WG);
  bf16* qb  = (bf16*)(ws + OFF_Q);
  bf16* kb  = (bf16*)(ws + OFF_K);
  bf16* vb  = (bf16*)(ws + OFF_V);
  bf16* ab  = (bf16*)(ws + OFF_A);
  float* out = (float*)d_out;

  auto blocks = [](long n) { return (unsigned)((n + 255) / 256); };

  // --- prep ---
  k_prep_wproj<<<blocks(C_ * 320), 256, 0, stream>>>(Wx, Wxg, wA);
  k_cvt<<<blocks((long)C_ * KK), 256, 0, stream>>>(Wq, wQ, C_ * KK);
  k_prep_wkv<<<blocks((long)2 * C_ * KK), 256, 0, stream>>>(Wk, Wv, wKV);
  k_prep_wgate<<<blocks((long)G_ * 128 * KG), 256, 0, stream>>>(
      Wxh_i, Wxh_f, Wxh_g, Wxh_o, Wa_i, Wa_f, Wa_g, Wa_o, wG);
  k_pack_bproj<<<blocks((long)N_ * HW * 320), 256, 0, stream>>>(inp, h_prev, Bp);
  k_pack_xhh<<<blocks((long)N_ * C_ * HW), 256, 0, stream>>>(h_prev, xh);

  // --- projection GEMM -> x-half of xh ---
  k_proj<<<(C_ / 64) * (N_ * HW / 64), 256, 0, stream>>>(wA, Bp, xh);

  // --- grouped convs ---
  k_gconv_q <<<N_ * G_ * (HW / 64), 256, 0, stream>>>(wQ, xh, qb);
  k_gconv_kv<<<N_ * G_ * (D_ / 64), 256, 0, stream>>>(wKV, xh, kb, vb);

  // --- attention ---
  k_attn<<<N_ * G_ * (HW / 16), 256, ATTN_SMEM, stream>>>(qb, kb, vb, tau, ab);

  // --- fused gates + LSTM cell -> h_new ---
  k_gates<<<N_ * G_ * (HW / 64), 256, 0, stream>>>(wG, xh, ab, c_prev,
                                                   b_i, b_f, b_g, b_o, out);
}
